// LinearLocalAttention_16999480557597
// MI455X (gfx1250) — compile-verified, matrix operand order checked by reading, not run on hardware
//
#include <hip/hip_runtime.h>
#include <hip/hip_bf16.h>

// ---------------------------------------------------------------------------
// LinearLocalAttention on MI455X (gfx1250).
// Key identity: softmax over K sums to 1 and y_v has no K axis, so the
// reference output is exactly  y_v = v_w @ diff + v_b  where
//   diff[b, c*K+k, n] = y[b,c,idx[b,n,k]] - y[b,c,n].
// Pipeline: (1) v_w -> bf16, (2) gather diff -> bf16 (per batch, reused ws),
// (3) WMMA bf16 GEMM with fp32 accumulation.
// This round: column-major B tile in LDS so every WMMA fragment load is a
// pair of 16B ds_load_b128 (no scalar ds_load_u16 / v_mov_b16 packing),
// plus global_prefetch of the next K-step.
// ---------------------------------------------------------------------------

typedef __bf16 bf16_t;
typedef __attribute__((ext_vector_type(16))) __bf16 v16bf;
typedef __attribute__((ext_vector_type(8)))  __bf16 v8bf;
typedef __attribute__((ext_vector_type(8)))  float  v8f;

#define BB 2
#define CC 256
#define NN 8192
#define KG 16
#define CK 4096   // CC * KG

// ---------------- weight conversion ----------------
__global__ void cvt_w_kernel(const float* __restrict__ w, bf16_t* __restrict__ w16, int n) {
    int i = blockIdx.x * blockDim.x + threadIdx.x;
    if (i < n) w16[i] = (bf16_t)w[i];
}

// ---------------- gather: diff16[(c*16+k)*NN + n] = y[c, idx[n,k]] - y[c,n] ----
__global__ __launch_bounds__(256)
void gather_diff_kernel(const float* __restrict__ y,      // batch base (CC x NN)
                        const int*   __restrict__ idx,    // batch base (NN x KG)
                        bf16_t*      __restrict__ diff16) // (CK x NN)
{
    const int n  = blockIdx.x * 256 + threadIdx.x;
    const int cg = blockIdx.y;                 // 16 channel groups of 16
    const int* ip = idx + (size_t)n * KG;
    int id[KG];
#pragma unroll
    for (int k = 0; k < KG; ++k) id[k] = ip[k];

#pragma unroll 4
    for (int cc = 0; cc < 16; ++cc) {
        const int c = cg * 16 + cc;
        const float* yr = y + (size_t)c * NN;
        const float ctr = yr[n];
#pragma unroll
        for (int k = 0; k < KG; ++k) {
            float v = yr[id[k]] - ctr;
            diff16[(size_t)(c * KG + k) * NN + n] = (bf16_t)v;
        }
    }
}

// ---------------- WMMA GEMM: out(256 x 8192) = Wv16(256 x 4096) @ diff16 + vb --
#define MT 128
#define NT 128
#define KT 32
#define ASTR 40    // 32 + 8 pad halves -> 80B rows, 16B aligned
#define BKS  40    // column-major B: 32 K-halves + 8 pad -> 80B cols, 16B aligned

__global__ __launch_bounds__(256)
void gemm_vw_kernel(const bf16_t* __restrict__ A,    // Wv16, 256 x 4096
                    const bf16_t* __restrict__ Bm,   // diff16, 4096 x 8192
                    const float*  __restrict__ bias, // v_b, 256
                    float*        __restrict__ out)  // 256 x 8192 (batch base)
{
    __shared__ alignas(16) bf16_t Asm[MT * ASTR];  // [row][k]
    __shared__ alignas(16) bf16_t Bsm[NT * BKS];   // [col][k]  (column-major tile)

    const int t    = threadIdx.x;
    const int lane = t & 31;
    const int wave = t >> 5;            // 0..7
    const int lh   = lane & 15;
    const int hs   = lane >> 4;         // half-wave select
    const int wm   = (wave >> 2) * 64;  // wave M offset within tile (0,64)
    const int wn   = (wave & 3) * 32;   // wave N offset within tile (0..96)
    const int m0   = blockIdx.y * MT;
    const int n0   = blockIdx.x * NT;

    // global-load assignments
    const int ar = t >> 1, ak = (t & 1) * 16;   // A: row 0..127, col base 0/16
    const int bkk = (t >> 4) * 2;               // B: even K row 0..30
    const int bcb = (t & 15) * 8;               // B: col base 0..120 (8 cols)

    v8f acc[4][2];
#pragma unroll
    for (int fm = 0; fm < 4; ++fm)
#pragma unroll
        for (int fn = 0; fn < 2; ++fn) acc[fm][fn] = (v8f){0.f,0.f,0.f,0.f,0.f,0.f,0.f,0.f};

    for (int kt = 0; kt < CK; kt += KT) {
        // ---- global fetches for this K-step
        const uint4* ap = (const uint4*)(A + (size_t)(m0 + ar) * CK + kt + ak);
        uint4 a0 = ap[0], a1 = ap[1];

        union { uint4 v; unsigned short h[8]; } r0, r1;
        r0.v = *(const uint4*)(Bm + (size_t)(kt + bkk)     * NN + n0 + bcb);
        r1.v = *(const uint4*)(Bm + (size_t)(kt + bkk + 1) * NN + n0 + bcb);

        // prefetch next K-step (global_prefetch_b8, speculative)
        if (kt + KT < CK) {
            __builtin_prefetch(A  + (size_t)(m0 + ar) * CK + kt + KT + ak, 0, 1);
            __builtin_prefetch(Bm + (size_t)(kt + KT + bkk) * NN + n0 + bcb, 0, 1);
        }

        __syncthreads();
        // A tile: row-major [row][k]
        *(uint4*)&Asm[ar * ASTR + ak]     = a0;
        *(uint4*)&Asm[ar * ASTR + ak + 8] = a1;
        // B tile: transpose into column-major [col][k]; pack (k,k+1) pairs -> b32
#pragma unroll
        for (int j = 0; j < 8; ++j) {
            unsigned pk = (unsigned)r0.h[j] | ((unsigned)r1.h[j] << 16);
            *(unsigned*)&Bsm[(bcb + j) * BKS + bkk] = pk;
        }
        __syncthreads();

        // ---- fragment loads: element i of lane -> K = i + (i&8) + 8*hs
        // both chunks are 16B-contiguous in LDS -> ds_load_b128
        v16bf af[4], bfr[2];
#pragma unroll
        for (int fm = 0; fm < 4; ++fm) {
            const int row = wm + fm * 16 + lh;
            v8bf c0 = *(const v8bf*)&Asm[row * ASTR + 8 * hs];
            v8bf c1 = *(const v8bf*)&Asm[row * ASTR + 16 + 8 * hs];
            af[fm] = __builtin_shufflevector(c0, c1, 0,1,2,3,4,5,6,7,8,9,10,11,12,13,14,15);
        }
#pragma unroll
        for (int fn = 0; fn < 2; ++fn) {
            const int col = wn + fn * 16 + lh;
            v8bf c0 = *(const v8bf*)&Bsm[col * BKS + 8 * hs];
            v8bf c1 = *(const v8bf*)&Bsm[col * BKS + 16 + 8 * hs];
            bfr[fn] = __builtin_shufflevector(c0, c1, 0,1,2,3,4,5,6,7,8,9,10,11,12,13,14,15);
        }

#pragma unroll
        for (int fm = 0; fm < 4; ++fm)
#pragma unroll
            for (int fn = 0; fn < 2; ++fn)
                acc[fm][fn] = __builtin_amdgcn_wmma_f32_16x16x32_bf16(
                    /*neg_a=*/false, af[fm], /*neg_b=*/false, bfr[fn],
                    /*c_mod=*/(short)0, acc[fm][fn],
                    /*reuse_a=*/false, /*reuse_b=*/false);
    }

    // epilogue: D layout -> M = r + 8*hs, N = lh
#pragma unroll
    for (int fm = 0; fm < 4; ++fm) {
#pragma unroll
        for (int fn = 0; fn < 2; ++fn) {
            const int colg = n0 + wn + fn * 16 + lh;
#pragma unroll
            for (int r = 0; r < 8; ++r) {
                const int rowg = m0 + wm + fm * 16 + r + 8 * hs;
                out[(size_t)rowg * NN + colg] = acc[fm][fn][r] + bias[rowg];
            }
        }
    }
}

// ---------------------------------------------------------------------------
extern "C" void kernel_launch(void* const* d_in, const int* in_sizes, int n_in,
                              void* d_out, int out_size, void* d_ws, size_t ws_size,
                              hipStream_t stream) {
    (void)out_size; (void)ws_size;
    const float* y   = (const float*)d_in[1];
    const int*   idx = (const int*)d_in[3];

    // locate v_w (unique 256*4096 tensor among params) and adjacent v_b (256)
    int iw = -1;
    for (int i = 4; i < n_in; ++i)
        if (in_sizes[i] == CC * CK) { iw = i; break; }
    const float* vw;
    const float* vb;
    if (iw >= 0) {
        vw = (const float*)d_in[iw];
        if (iw + 1 < n_in && in_sizes[iw + 1] == CC) vb = (const float*)d_in[iw + 1];
        else                                         vb = (const float*)d_in[iw - 1];
    } else {
        vw = (const float*)d_in[8];
        vb = (const float*)d_in[9];
    }

    float* out = (float*)d_out;

    // workspace: [Wv16: 2MB][diff16: 67MB, reused per batch]
    bf16_t* Wv16   = (bf16_t*)d_ws;
    bf16_t* diff16 = (bf16_t*)((char*)d_ws + (size_t)CC * CK * sizeof(bf16_t));

    cvt_w_kernel<<<(CC * CK + 255) / 256, 256, 0, stream>>>(vw, Wv16, CC * CK);

    for (int b = 0; b < BB; ++b) {
        gather_diff_kernel<<<dim3(NN / 256, 16), 256, 0, stream>>>(
            y + (size_t)b * CC * NN, idx + (size_t)b * NN * KG, diff16);
        gemm_vw_kernel<<<dim3(NN / NT, CC / MT), 256, 0, stream>>>(
            Wv16, diff16, vb, out + (size_t)b * CC * NN);
    }
}
